// GATLinkPredictor_63771674411416
// MI455X (gfx1250) — compile-verified, matrix operand order checked by reading, not run on hardware
//
#include <hip/hip_runtime.h>
#include <hip/hip_bf16.h>

// ---------------------------------------------------------------------------
// GAT link predictor for MI455X (gfx1250, wave32, WMMA).
//  * All GEMMs: v_wmma_f32_16x16x32_bf16, fp32 accumulate.
//  * Activations/weights pre-converted to bf16 so A fragments are two
//    global_load_b128 per lane and B fragments are two ds_load_b128 from an
//    LDS-staged transposed-weight panel (shared by all 4 waves of a block).
//  * Edge softmax/aggregation: coalesced HBM-bound scatter passes (the real
//    roofline cost: ~3 GB of gather/atomic traffic per GAT layer).
// ---------------------------------------------------------------------------

#define N_NODES 50000
#define N_EDGES 1000000
#define N_EVAL  100000
#define IN_DIM  256
#define HID     64
#define HEADS   4
#define FEAT    (HEADS * HID)   // 256
#define NEG_SLOPE 0.2f

typedef __attribute__((ext_vector_type(16))) __bf16 v16bf;
typedef __attribute__((ext_vector_type(8)))  __bf16 v8bf;
typedef __attribute__((ext_vector_type(8)))  float  v8f;

// Assemble a 16x32 bf16 WMMA fragment from two contiguous 8-element runs.
// ISA 7.12.2: lane = 16*h + m, VGPR j holds K = kk + (j/4)*16 + h*8 + (j%4)*2.
// => elements 0..7 = [kk+h*8 .. +7], elements 8..15 = [kk+16+h*8 .. +7].
__device__ __forceinline__ v16bf frag_from_runs(v8bf lo, v8bf hi) {
    return __builtin_shufflevector(lo, hi, 0, 1, 2, 3, 4, 5, 6, 7,
                                           8, 9, 10, 11, 12, 13, 14, 15);
}

// ---------------- Generic GEMM ---------------------------------------------
// out[M,N] = act(A[M,K] @ Wt[N,K]^T + bias).  A,Wt bf16 row-major.
// One wave per 16-row strip; NT col tiles of 16; Wt panel staged in LDS.
// M%16==0, K%32==0 assumed (true for all shapes here).
template <int NT, bool RELU, bool BIAS, bool OUT_BF16>
__global__ void gemm_wmma(const __bf16* __restrict__ A,
                          const __bf16* __restrict__ Wt,
                          const float* __restrict__ bias,
                          void* __restrict__ outp,
                          int M, int K, int N) {
    extern __shared__ __align__(16) char smem_raw[];
    __bf16* sB = (__bf16*)smem_raw;

    const int lane = threadIdx.x & 31;
    const int wave = threadIdx.x >> 5;
    const int row0 = (blockIdx.x * (blockDim.x >> 5) + wave) * 16;
    const int col0 = blockIdx.y * NT * 16;

    // Cooperative stage of the weight panel: rows [col0, col0+NT*8) of Wt,
    // a contiguous NT*8*K bf16 block.  16-byte chunks.
    {
        const uint4* gsrc = (const uint4*)(Wt + (size_t)col0 * K);
        uint4* sdst = (uint4*)sB;
        const int chunks = NT * K;            // NT*8*K*2 / 16 (half the panel)
        for (int c = threadIdx.x; c < chunks; c += blockDim.x) sdst[c] = gsrc[c];
    }
    __syncthreads();

    if (row0 >= M) return;

    const int m = lane & 15;
    const int h = lane >> 4;

    v8f acc[NT];
#pragma unroll
    for (int t = 0; t < NT; ++t) acc[t] = {};

    const __bf16* arow = A + (size_t)(row0 + m) * K;
    const v8bf* sBv = (const v8bf*)sB;
    for (int kk = 0; kk < K; kk += 32) {
        const v8bf alo = *(const v8bf*)(arow + kk + h * 8);
        const v8bf ahi = *(const v8bf*)(arow + kk + 16 + h * 8);
        const v16bf a = frag_from_runs(alo, ahi);
#pragma unroll
        for (int t = 0; t < NT; ++t) {
            const int rl = t * 16 + m;        // local Wt row (lane&15 == n)
            const v8bf blo = sBv[((size_t)rl * K + kk + h * 8) >> 3];
            const v8bf bhi = sBv[((size_t)rl * K + kk + 16 + h * 8) >> 3];
            const v16bf b = frag_from_runs(blo, bhi);
            acc[t] = __builtin_amdgcn_wmma_f32_16x16x32_bf16(
                false, a, false, b, (short)0, acc[t], false, false);
        }
    }

    // C/D layout: VGPR r -> M = r (+8 for upper lane half), lane&15 -> N
    const int mb = (lane >> 4) * 8;
#pragma unroll
    for (int t = 0; t < NT; ++t) {
        const int c = col0 + 16 * t + m;
#pragma unroll
        for (int r = 0; r < 8; ++r) {
            float v = acc[t][r];
            if (BIAS) v += bias[c];
            if (RELU) v = v > 0.f ? v : 0.f;
            const size_t idx = (size_t)(row0 + mb + r) * N + c;
            if (OUT_BF16) ((__bf16*)outp)[idx] = (__bf16)v;
            else          ((float*)outp)[idx] = v;
        }
    }
}

// ---------------- lp1 GEMM with fused pair gather --------------------------
// A row i = concat(EMBh[sidx[i]], EMBh[didx[i]])  (bf16, K = 512), N = 64.
__global__ void gemm_wmma_pair(const __bf16* __restrict__ EMBh,
                               const long long* __restrict__ sidx,
                               const long long* __restrict__ didx,
                               const __bf16* __restrict__ Wt,  // [64,512] bf16
                               const float* __restrict__ bias,
                               __bf16* __restrict__ out, int M) {
    constexpr int K = 2 * FEAT;   // 512
    constexpr int N = HID;        // 64
    constexpr int NT = 4;
    extern __shared__ __align__(16) char smem_raw[];
    __bf16* sB = (__bf16*)smem_raw;

    const int lane = threadIdx.x & 31;
    const int wave = threadIdx.x >> 5;
    const int row0 = (blockIdx.x * (blockDim.x >> 5) + wave) * 16;

    {   // stage first 32 rows of the 64x512 bf16 weight panel (32 KB)
        const uint4* gsrc = (const uint4*)Wt;
        uint4* sdst = (uint4*)sB;
        for (int c = threadIdx.x; c < NT * K; c += blockDim.x) sdst[c] = gsrc[c];
    }
    __syncthreads();

    if (row0 >= M) return;

    const int m = lane & 15;
    const int h = lane >> 4;
    const __bf16* srow = EMBh + (size_t)sidx[row0 + m] * FEAT;
    const __bf16* drow = EMBh + (size_t)didx[row0 + m] * FEAT;

    v8f acc[NT];
#pragma unroll
    for (int t = 0; t < NT; ++t) acc[t] = {};

    const v8bf* sBv = (const v8bf*)sB;
    for (int kk = 0; kk < K; kk += 32) {
        // kk..kk+31 lies entirely in one half (256-boundary is a multiple of 32)
        const __bf16* base = (kk < FEAT) ? (srow + kk) : (drow + kk - FEAT);
        const v16bf a = frag_from_runs(*(const v8bf*)(base + h * 8),
                                       *(const v8bf*)(base + 16 + h * 8));
#pragma unroll
        for (int t = 0; t < NT; ++t) {
            const int rl = t * 16 + m;
            const v8bf blo = sBv[((size_t)rl * K + kk + h * 8) >> 3];
            const v8bf bhi = sBv[((size_t)rl * K + kk + 16 + h * 8) >> 3];
            acc[t] = __builtin_amdgcn_wmma_f32_16x16x32_bf16(
                false, a, false, frag_from_runs(blo, bhi), (short)0, acc[t],
                false, false);
        }
    }

    const int mb = (lane >> 4) * 8;
#pragma unroll
    for (int t = 0; t < NT; ++t) {
        const int c = 16 * t + m;
#pragma unroll
        for (int r = 0; r < 8; ++r) {
            float v = acc[t][r] + bias[c];
            v = v > 0.f ? v : 0.f;
            out[(size_t)(row0 + mb + r) * N + c] = (__bf16)v;
        }
    }
}

// ---------------- prep / elementwise kernels -------------------------------
__global__ void cvt_f32_bf16(const float* __restrict__ src,
                             __bf16* __restrict__ dst, size_t n) {
    size_t i = (size_t)blockIdx.x * blockDim.x + threadIdx.x;
    if (i < n) dst[i] = (__bf16)src[i];
}

// W[K,N] f32 -> Wt[N,K] bf16
__global__ void transpose_w(const float* __restrict__ W,
                            __bf16* __restrict__ Wt, int K, int N) {
    int i = blockIdx.x * blockDim.x + threadIdx.x;   // n*K + k
    if (i >= K * N) return;
    const int n = i / K, k = i - n * K;
    Wt[i] = (__bf16)W[(size_t)k * N + n];
}

__global__ void fill_kernel(float* __restrict__ p, size_t n, float v) {
    size_t i = (size_t)blockIdx.x * blockDim.x + threadIdx.x;
    if (i < n) p[i] = v;
}

// a_s[n,h] = <T[n,h,:], att_src[h,:]>, same for a_d
__global__ void att_logits(const float* __restrict__ T,
                           const float* __restrict__ att_s,
                           const float* __restrict__ att_d,
                           float* __restrict__ as_, float* __restrict__ ad_,
                           int nNodes) {
    int t = blockIdx.x * blockDim.x + threadIdx.x;
    if (t >= nNodes * HEADS) return;
    const int hh = t & (HEADS - 1);
    const int n  = t >> 2;
    const float* row = T + (size_t)n * FEAT + hh * HID;
    const float* ws  = att_s + hh * HID;
    const float* wd  = att_d + hh * HID;
    float ss = 0.f, dd = 0.f;
#pragma unroll 4
    for (int c = 0; c < HID; ++c) { ss += row[c] * ws[c]; dd += row[c] * wd[c]; }
    as_[t] = ss;
    ad_[t] = dd;
}

__device__ __forceinline__ void atomicMaxF(float* addr, float val) {
    if (val >= 0.f) atomicMax((int*)addr, __float_as_int(val));
    else            atomicMin((unsigned int*)addr, __float_as_uint(val));
}

__global__ void edge_logit_max(const long long* __restrict__ ei,
                               const float* __restrict__ as_,
                               const float* __restrict__ ad_,
                               float* __restrict__ ee, float* __restrict__ mx) {
    int t = blockIdx.x * blockDim.x + threadIdx.x;
    if (t >= N_EDGES * HEADS) return;
    const int hh = t & (HEADS - 1);
    const int e  = t >> 2;
    const long long s = ei[e];
    const long long d = ei[N_EDGES + e];
    float v = as_[s * HEADS + hh] + ad_[d * HEADS + hh];
    v = v > 0.f ? v : NEG_SLOPE * v;
    ee[t] = v;
    atomicMaxF(&mx[d * HEADS + hh], v);
}

__global__ void edge_exp_sum(const long long* __restrict__ ei,
                             const float* __restrict__ mx,
                             float* __restrict__ ee, float* __restrict__ sm) {
    int t = blockIdx.x * blockDim.x + threadIdx.x;
    if (t >= N_EDGES * HEADS) return;
    const int hh = t & (HEADS - 1);
    const int e  = t >> 2;
    const long long d = ei[N_EDGES + e];
    const float v = __expf(ee[t] - mx[d * HEADS + hh]);
    ee[t] = v;
    atomicAdd(&sm[d * HEADS + hh], v);
}

// One wave per edge: AGG[dst] += T[src] * alpha   (256 features, 8 per lane)
__global__ void edge_aggregate(const long long* __restrict__ ei,
                               const float* __restrict__ T,
                               const float* __restrict__ ee,
                               const float* __restrict__ sm,
                               float* __restrict__ AGG) {
    const int lane = threadIdx.x & 31;
    const int wv   = (blockIdx.x * blockDim.x + threadIdx.x) >> 5;
    if (wv >= N_EDGES) return;
    const long long s = ei[wv];
    const long long d = ei[N_EDGES + wv];
    const float* src = T + (size_t)s * FEAT;
    float*       dst = AGG + (size_t)d * FEAT;
#pragma unroll
    for (int i = 0; i < FEAT / 32; ++i) {
        const int f  = i * 32 + lane;
        const int hh = f >> 6;
        const float alpha = ee[(size_t)wv * HEADS + hh] /
                            (sm[d * HEADS + hh] + 1e-16f);
        atomicAdd(&dst[f], src[f] * alpha);
    }
}

__global__ void add_bias(float* __restrict__ p, const float* __restrict__ bias,
                         int rows, int cols) {
    size_t i = (size_t)blockIdx.x * blockDim.x + threadIdx.x;
    if (i >= (size_t)rows * cols) return;
    p[i] += bias[i % cols];
}

// scores[i] = <z2[i,:32], lp3_w> + lp3_b
__global__ void lp3_kernel(const float* __restrict__ z2,
                           const float* __restrict__ w,
                           const float* __restrict__ b,
                           float* __restrict__ out, int M) {
    int i = blockIdx.x * blockDim.x + threadIdx.x;
    if (i >= M) return;
    const float* row = z2 + (size_t)i * 32;
    float acc = b[0];
#pragma unroll
    for (int j = 0; j < 32; ++j) acc += row[j] * w[j];
    out[i] = acc;
}

// ---------------------------------------------------------------------------
extern "C" void kernel_launch(void* const* d_in, const int* in_sizes, int n_in,
                              void* d_out, int out_size, void* d_ws, size_t ws_size,
                              hipStream_t stream) {
    const float*     x     = (const float*)d_in[0];
    const long long* ei    = (const long long*)d_in[1];
    const long long* esrc  = (const long long*)d_in[2];
    const long long* edst  = (const long long*)d_in[3];
    const float*     enc_w = (const float*)d_in[4];
    const float*     enc_b = (const float*)d_in[5];
    const float*     g1_w  = (const float*)d_in[6];
    const float*     g1_as = (const float*)d_in[7];
    const float*     g1_ad = (const float*)d_in[8];
    const float*     g1_b  = (const float*)d_in[9];
    const float*     g2_w  = (const float*)d_in[10];
    const float*     g2_as = (const float*)d_in[11];
    const float*     g2_ad = (const float*)d_in[12];
    const float*     g2_b  = (const float*)d_in[13];
    const float*     lp1_w = (const float*)d_in[14];
    const float*     lp1_b = (const float*)d_in[15];
    const float*     lp2_w = (const float*)d_in[16];
    const float*     lp2_b = (const float*)d_in[17];
    const float*     lp3_w = (const float*)d_in[18];
    const float*     lp3_b = (const float*)d_in[19];
    float*           out   = (float*)d_out;
    (void)in_sizes; (void)n_in; (void)out_size; (void)ws_size;

    // ---- workspace layout (256B-aligned byte offsets) ----
    char* ws = (char*)d_ws;
    size_t off = 0;
    auto alloc = [&](size_t bytes) {
        char* p = ws + off;
        off += (bytes + 255) & ~(size_t)255;
        return p;
    };
    __bf16* Xh    = (__bf16*)alloc((size_t)N_NODES * IN_DIM * 2);
    __bf16* H0h   = (__bf16*)alloc((size_t)N_NODES * HID * 2);
    float*  T     = (float*) alloc((size_t)N_NODES * FEAT * 4);
    float*  AGG   = (float*) alloc((size_t)N_NODES * FEAT * 4);
    __bf16* AGGh  = (__bf16*)alloc((size_t)N_NODES * FEAT * 2);
    float*  AS    = (float*) alloc((size_t)N_NODES * HEADS * 4);
    float*  AD    = (float*) alloc((size_t)N_NODES * HEADS * 4);
    float*  MX    = (float*) alloc((size_t)N_NODES * HEADS * 4);
    float*  SM    = (float*) alloc((size_t)N_NODES * HEADS * 4);
    float*  EE    = (float*) alloc((size_t)N_EDGES * HEADS * 4);
    __bf16* Z1h   = (__bf16*)alloc((size_t)N_EVAL * HID * 2);
    float*  Z2    = (float*) alloc((size_t)N_EVAL * 32 * 4);
    __bf16* encWt = (__bf16*)alloc((size_t)IN_DIM * HID * 2);   // [64,256]
    __bf16* g1Wt  = (__bf16*)alloc((size_t)HID * FEAT * 2);     // [256,64]
    __bf16* g2Wt  = (__bf16*)alloc((size_t)FEAT * FEAT * 2);    // [256,256]
    __bf16* lp1Wt = (__bf16*)alloc((size_t)2 * FEAT * HID * 2); // [64,512]
    __bf16* lp2Wt = (__bf16*)alloc((size_t)HID * 32 * 2);       // [32,64]

    const int WPB = 4, GEMM_BLK = WPB * 32;
    auto ggrid = [&](int M) { return (M / 16 + WPB - 1) / WPB; };
    auto ldsB  = [&](int NT, int K) { return (size_t)NT * 16 * K * 2; };

    const int nodeHeads = N_NODES * HEADS;
    const int edgeHeads = N_EDGES * HEADS;
    const size_t featElems = (size_t)N_NODES * FEAT;
    const size_t xElems = (size_t)N_NODES * IN_DIM;

    // ---- prep: transpose weights to [N,K] bf16, convert x ----
    transpose_w<<<(IN_DIM * HID + 255) / 256, 256, 0, stream>>>(enc_w, encWt, IN_DIM, HID);
    transpose_w<<<(HID * FEAT + 255) / 256, 256, 0, stream>>>(g1_w, g1Wt, HID, FEAT);
    transpose_w<<<(FEAT * FEAT + 255) / 256, 256, 0, stream>>>(g2_w, g2Wt, FEAT, FEAT);
    transpose_w<<<(2 * FEAT * HID + 255) / 256, 256, 0, stream>>>(lp1_w, lp1Wt, 2 * FEAT, HID);
    transpose_w<<<(HID * 32 + 255) / 256, 256, 0, stream>>>(lp2_w, lp2Wt, HID, 32);
    cvt_f32_bf16<<<(unsigned)((xElems + 255) / 256), 256, 0, stream>>>(x, Xh, xElems);

    // 1) encoder: H0h = relu(x @ enc_w + enc_b)  [bf16 out]
    gemm_wmma<4, true, true, true><<<dim3(ggrid(N_NODES), 1), GEMM_BLK,
        ldsB(4, IN_DIM), stream>>>(Xh, encWt, enc_b, H0h, N_NODES, IN_DIM, HID);

    // ---- GAT layer 1 ----
    gemm_wmma<4, false, false, false><<<dim3(ggrid(N_NODES), FEAT / 64), GEMM_BLK,
        ldsB(4, HID), stream>>>(H0h, g1Wt, nullptr, T, N_NODES, HID, FEAT);
    att_logits<<<(nodeHeads + 255) / 256, 256, 0, stream>>>(T, g1_as, g1_ad, AS, AD, N_NODES);
    fill_kernel<<<(nodeHeads + 255) / 256, 256, 0, stream>>>(MX, nodeHeads, -__builtin_huge_valf());
    fill_kernel<<<(nodeHeads + 255) / 256, 256, 0, stream>>>(SM, nodeHeads, 0.f);
    fill_kernel<<<(unsigned)((featElems + 255) / 256), 256, 0, stream>>>(AGG, featElems, 0.f);
    edge_logit_max<<<(edgeHeads + 255) / 256, 256, 0, stream>>>(ei, AS, AD, EE, MX);
    edge_exp_sum<<<(edgeHeads + 255) / 256, 256, 0, stream>>>(ei, MX, EE, SM);
    edge_aggregate<<<(N_EDGES * 32 + 255) / 256, 256, 0, stream>>>(ei, T, EE, SM, AGG);
    add_bias<<<(unsigned)((featElems + 255) / 256), 256, 0, stream>>>(AGG, g1_b, N_NODES, FEAT);
    cvt_f32_bf16<<<(unsigned)((featElems + 255) / 256), 256, 0, stream>>>(AGG, AGGh, featElems);

    // ---- GAT layer 2 ----
    gemm_wmma<4, false, false, false><<<dim3(ggrid(N_NODES), FEAT / 64), GEMM_BLK,
        ldsB(4, FEAT), stream>>>(AGGh, g2Wt, nullptr, T, N_NODES, FEAT, FEAT);
    att_logits<<<(nodeHeads + 255) / 256, 256, 0, stream>>>(T, g2_as, g2_ad, AS, AD, N_NODES);
    fill_kernel<<<(nodeHeads + 255) / 256, 256, 0, stream>>>(MX, nodeHeads, -__builtin_huge_valf());
    fill_kernel<<<(nodeHeads + 255) / 256, 256, 0, stream>>>(SM, nodeHeads, 0.f);
    fill_kernel<<<(unsigned)((featElems + 255) / 256), 256, 0, stream>>>(AGG, featElems, 0.f);
    edge_logit_max<<<(edgeHeads + 255) / 256, 256, 0, stream>>>(ei, AS, AD, EE, MX);
    edge_exp_sum<<<(edgeHeads + 255) / 256, 256, 0, stream>>>(ei, MX, EE, SM);
    edge_aggregate<<<(N_EDGES * 32 + 255) / 256, 256, 0, stream>>>(ei, T, EE, SM, AGG);
    add_bias<<<(unsigned)((featElems + 255) / 256), 256, 0, stream>>>(AGG, g2_b, N_NODES, FEAT);
    cvt_f32_bf16<<<(unsigned)((featElems + 255) / 256), 256, 0, stream>>>(AGG, AGGh, featElems);

    // ---- link predictor ----
    gemm_wmma_pair<<<dim3(ggrid(N_EVAL), 1), GEMM_BLK,
        ldsB(4, 2 * FEAT), stream>>>(AGGh, esrc, edst, lp1Wt, lp1_b, Z1h, N_EVAL);
    gemm_wmma<2, true, true, false><<<dim3(ggrid(N_EVAL), 1), GEMM_BLK,
        ldsB(2, HID), stream>>>(Z1h, lp2Wt, lp2_b, Z2, N_EVAL, HID, 32);
    lp3_kernel<<<(N_EVAL + 255) / 256, 256, 0, stream>>>(Z2, lp3_w, lp3_b, out, N_EVAL);
}
